// RBFNetwork_58171037057369
// MI455X (gfx1250) — matrix operand order repaired and checked by reading, not compiled, not run on hardware
//
#include <hip/hip_runtime.h>

typedef __attribute__((ext_vector_type(2)))  float  v2f;
typedef __attribute__((ext_vector_type(4)))  float  v4f;
typedef __attribute__((ext_vector_type(8)))  float  v8f;
typedef __attribute__((ext_vector_type(8)))  __bf16 v8bf;
typedef __attribute__((ext_vector_type(16))) __bf16 v16bf;

#define BN 131072
#define DK 64
#define HN 512
#define ON 16

// Block = 256 threads = 8 waves; each wave owns 16 rows of x; block owns 128 rows.
__global__ __launch_bounds__(256) void rbf_fused_kernel(
    const float* __restrict__ x, const float* __restrict__ centroid,
    const float* __restrict__ sigma, const float* __restrict__ weight,
    const float* __restrict__ bias, float* __restrict__ out)
{
  // LDS: ~28 KB total
  __shared__ __bf16 s_wB[16 * 32 * 16];   // weight pre-packed as bf16 WMMA B-frags: [ks][lane][16 K-halves]
  __shared__ __bf16 s_hid[8][16][32];     // per-wave hidden staging for K=32 block (transpose C->A layout)
  __shared__ float  s_csq[HN];            // ||c_h||^2
  __shared__ float  s_scl[HN];            // -1/(2*sigma_h^2)
  __shared__ float  s_bias[ON];

  const int tid = threadIdx.x;

  // ---- Phase 1: cooperative precompute into LDS ----
  for (int h = tid; h < HN; h += 256) {
    const float* c = centroid + h * DK;
    float s = 0.f;
    #pragma unroll
    for (int d = 0; d < DK; d += 4) {
      v4f t = *(const v4f*)(c + d);
      s += t.x * t.x + t.y * t.y + t.z * t.z + t.w * t.w;
    }
    s_csq[h] = s;
    float sg = sigma[h];
    s_scl[h] = -0.5f / (sg * sg);
  }
  // Pack weight into the exact B-fragment layout of v_wmma_f32_16x16x32_bf16:
  // lane (0..31): N = lane&15, K = ks*32 + 8*(lane>>4) + (i&7) + 16*(i>>3) for element i.
  for (int idx = tid; idx < 16 * 32 * 16; idx += 256) {
    int i = idx & 15, lane = (idx >> 4) & 31, ks = idx >> 9;
    int h = ks * 32 + ((i >> 3) << 4) + ((lane >> 4) << 3) + (i & 7);
    int n = lane & 15;
    s_wB[idx] = (__bf16)weight[h * ON + n];
  }
  if (tid < ON) s_bias[tid] = bias[tid];
  __syncthreads();

  const int w    = tid >> 5;
  const int l    = tid & 31;
  const int half = l >> 4;
  const int n16  = l & 15;
  const long rowbase = (long)blockIdx.x * 128 + w * 16;

  // ---- Hoist x A-fragments (read x exactly once per wave tile) ----
  // f32 WMMA 16x16x4 A layout: lanes 0-15 row M=lane K={0,1}; lanes 16-31 same rows K={2,3}.
  const float* xrow = x + (rowbase + n16) * DK;
  v2f afr[16];
  #pragma unroll
  for (int k = 0; k < 16; ++k)
    afr[k] = *(const v2f*)(xrow + 4 * k + 2 * half);

  // ||x_row||^2 : each lane sums half a row, combine across half-waves.
  float xp = 0.f;
  #pragma unroll
  for (int d = 0; d < 8; ++d) {
    v4f t = *(const v4f*)(xrow + half * 32 + 4 * d);
    xp += t.x * t.x + t.y * t.y + t.z * t.z + t.w * t.w;
  }
  xp += __shfl_xor(xp, 16);
  // xs[j] = ||x_{rowbase + j + 8*half}||^2 (matches C-layout row ownership)
  float xs[8];
  #pragma unroll
  for (int j = 0; j < 8; ++j)
    xs[j] = __shfl(xp, j + 8 * half);

  v8f acc2 = {};  // GEMM2 accumulator: out[16 rows x 16 outputs]

  // ---- Double-buffered centroid B-fragments: WMMA chain never waits on VMEM ----
  v2f bbuf[2][16];
  const float* cb0 = centroid + n16 * DK + 2 * half;
  #pragma unroll
  for (int k = 0; k < 16; ++k)
    bbuf[0][k] = *(const v2f*)(cb0 + 4 * k);

  #pragma unroll 2
  for (int ntile = 0; ntile < 32; ++ntile) {
    const int cur = ntile & 1, nxt = cur ^ 1;

    // Prefetch next centroid tile's 16 B-fragments (one load clause, overlaps WMMAs below).
    if (ntile < 31) {
      const float* cbn = cb0 + (ntile + 1) * (16 * DK);
      #pragma unroll
      for (int k = 0; k < 16; ++k)
        bbuf[nxt][k] = *(const v2f*)(cbn + 4 * k);
    }

    // ---- GEMM1: dot(x_tile, centroid_tile) via f32 WMMA, K = 64 in 16 steps ----
    v8f acc = {};
    #pragma unroll
    for (int k = 0; k < 16; ++k)
      acc = __builtin_amdgcn_wmma_f32_16x16x4_f32(
          false, afr[k], false, bbuf[cur][k], (short)0, acc, false, false);

    // ---- RBF activation in C layout: lane holds h = ntile*16 + n16, rows j+8*half ----
    const int hidx = ntile * 16 + n16;
    const float csq = s_csq[hidx];
    const float scl = s_scl[hidx];
    const int hc = (ntile & 1) * 16 + n16;
    #pragma unroll
    for (int j = 0; j < 8; ++j) {
      float sqd = xs[j] + csq - 2.0f * acc[j];
      float hid = __expf(sqd * scl);
      s_hid[w][j + 8 * half][hc] = (__bf16)hid;   // transpose staging (wave-private)
    }

    // ---- Every 2 ntiles: one bf16 WMMA K=32 step of GEMM2 ----
    if (ntile & 1) {
      const int ks = ntile >> 1;
      asm volatile("s_wait_dscnt 0" ::: "memory");  // wave-level RAW fence on LDS staging
      // A layout (16-bit 16x32): lanes0-15 M=lane, K 0..7 & 16..23; lanes16-31 K 8..15 & 24..31
      v8bf alo = *(const v8bf*)(&s_hid[w][n16][8 * half]);
      v8bf ahi = *(const v8bf*)(&s_hid[w][n16][8 * half + 16]);
      v16bf a2 = __builtin_shufflevector(alo, ahi,
          0, 1, 2, 3, 4, 5, 6, 7, 8, 9, 10, 11, 12, 13, 14, 15);
      v16bf b2 = *(const v16bf*)(&s_wB[(ks * 32 + l) * 16]);
      acc2 = __builtin_amdgcn_wmma_f32_16x16x32_bf16(
          false, a2, false, b2, (short)0, acc2, false, false);
    }
  }

  // ---- Epilogue: +bias, row-normalize over the 16 outputs, store ----
  const float bo = s_bias[n16];
  #pragma unroll
  for (int j = 0; j < 8; ++j) {
    float v = acc2[j] + bo;           // out[row, o=n16]
    float s = v;
    s += __shfl_xor(s, 1);
    s += __shfl_xor(s, 2);
    s += __shfl_xor(s, 4);
    s += __shfl_xor(s, 8);            // sum over the 16 outputs (stays in half-wave)
    const long row = rowbase + j + 8 * half;
    out[row * ON + n16] = v / s;
  }
}

extern "C" void kernel_launch(void* const* d_in, const int* in_sizes, int n_in,
                              void* d_out, int out_size, void* d_ws, size_t ws_size,
                              hipStream_t stream) {
  (void)in_sizes; (void)n_in; (void)out_size; (void)d_ws; (void)ws_size;
  const float* x        = (const float*)d_in[0];
  const float* centroid = (const float*)d_in[1];
  const float* sigma    = (const float*)d_in[2];
  const float* weight   = (const float*)d_in[3];
  const float* bias     = (const float*)d_in[4];
  float* out            = (float*)d_out;

  dim3 grid(BN / 128), block(256);
  rbf_fused_kernel<<<grid, block, 0, stream>>>(x, centroid, sigma, weight, bias, out);
}
